// QuantumLSTM_65481071399342
// MI455X (gfx1250) — matrix operand order repaired and checked by reading, not compiled, not run on hardware
//
#include <hip/hip_runtime.h>
#include <math.h>

// ---------------------------------------------------------------------------
// QuantumLSTM on gfx1250 (CDNA5, wave32, WMMA)
//   SEQ=512, BATCH=256, IN_DIM=512, HID=256, GATES=4*HID=1024
//
// Phase A (parallel): xz[t,b,4H] = x @ Wx^T + bias   (f16 WMMA, f32 acc)
//                     stored pre-swizzled in WMMA C/D register layout.
// Phase B (persistent, 32 blocks): 512 serial steps; per step a
//                     [256x256]@[256x1024] f16 WMMA GEMM (Wh in LDS),
//                     z=-sin(.), gates, c/h update; c lives in VGPRs.
//                     Next step's xz is register-prefetched before the
//                     device-wide barrier so HBM latency hides under the sync.
// ---------------------------------------------------------------------------

typedef __attribute__((ext_vector_type(16))) _Float16 v16h;
typedef __attribute__((ext_vector_type(8)))  _Float16 v8h;
typedef __attribute__((ext_vector_type(8)))  float    v8f;
typedef __attribute__((ext_vector_type(4)))  float    v4f;

#define SEQ    512
#define BATCH  256
#define IN_DIM 512
#define HID    256

// workspace layout (bytes)
#define WS_CTR   0                                   // 256 B barrier counter
#define WS_WX    256                                 // 1024*512 f16 = 1 MB
#define WS_WH    (WS_WX + 1024*512*2)                // 1024*256 f16 = 512 KB
#define WS_HBUF  (WS_WH + 1024*256*2)                // 2 * 256*256 f16 = 256 KB
#define WS_XZ    (WS_HBUF + 2*256*256*2)             // 512*256*1024 f32 = 512 MB

__device__ __forceinline__ float sigm(float z) {
    return 1.0f / (1.0f + __expf(-z));
}
// fast tanh: 2*sigmoid(2x)-1, v_exp_f32 path (precision on par with f16 GEMM)
__device__ __forceinline__ float ftanh(float x) {
    return 2.0f / (1.0f + __expf(-2.0f * x)) - 1.0f;
}

// ---------------------------------------------------------------------------
// Kernel 0: split+convert W[4][256][768] f32 -> Wx f16 [1024][512], Wh f16 [1024][256]
// ---------------------------------------------------------------------------
__global__ __launch_bounds__(256)
void qlstm_convw(const float* __restrict__ Wf, const float* __restrict__ Wi,
                 const float* __restrict__ Wg, const float* __restrict__ Wo,
                 _Float16* __restrict__ Wx, _Float16* __restrict__ Wh)
{
    int idx = blockIdx.x * 256 + threadIdx.x;
    if (idx >= 1024 * 768) return;
    int n = idx / 768;           // gate-major row 0..1023
    int c = idx % 768;
    int g = n >> 8;
    int r = n & 255;
    const float* W = (g == 0) ? Wf : (g == 1) ? Wi : (g == 2) ? Wg : Wo;
    float v = W[r * 768 + c];
    if (c < IN_DIM) Wx[(size_t)n * IN_DIM + c]            = (_Float16)v;
    else            Wh[(size_t)n * HID    + (c - IN_DIM)] = (_Float16)v;
}

// ---------------------------------------------------------------------------
// Phase A: xz = x @ Wx^T + b.   One wave per (m_tile, ji); 4 gate tiles each.
// Output pre-swizzled: xz[ patch*256 + lane*8 + reg ],
//   patch = ((t*4+g)*16 + ji)*16 + mi
// ---------------------------------------------------------------------------
__global__ __launch_bounds__(256)
void qlstm_xproj(const float* __restrict__ x,
                 const _Float16* __restrict__ Wx,
                 const float* __restrict__ bf, const float* __restrict__ bi,
                 const float* __restrict__ bg, const float* __restrict__ bo,
                 float* __restrict__ xz)
{
    const int lane = threadIdx.x & 31;
    const int wi   = threadIdx.x >> 5;
    const int wave = blockIdx.x * 8 + wi;          // 0..131071
    const int m_tile = wave >> 4;                  // 0..8191  (row tile of 16)
    const int ji     = wave & 15;                  // hidden tile 0..15
    const int half   = lane >> 4;                  // K-half selector
    const int ln     = lane & 15;
    const int mrow   = m_tile * 16 + ln;           // A row per lane

    v8f acc[4] = { {}, {}, {}, {} };
    const float* xrow = x + (size_t)mrow * IN_DIM;

    for (int k0 = 0; k0 < IN_DIM; k0 += 32) {
        // ---- A fragment (16x32 f16): lane half picks K octets per ISA layout
        const int c0 = k0 + half * 8;
        v4f xa0 = *reinterpret_cast<const v4f*>(xrow + c0);
        v4f xa1 = *reinterpret_cast<const v4f*>(xrow + c0 + 4);
        v4f xb0 = *reinterpret_cast<const v4f*>(xrow + c0 + 16);
        v4f xb1 = *reinterpret_cast<const v4f*>(xrow + c0 + 20);
        v16h a;
        #pragma unroll
        for (int e = 0; e < 4; ++e) {
            a[e]      = (_Float16)xa0[e];
            a[4 + e]  = (_Float16)xa1[e];
            a[8 + e]  = (_Float16)xb0[e];
            a[12 + e] = (_Float16)xb1[e];
        }
        // ---- 4 gate B fragments + WMMA
        #pragma unroll
        for (int g = 0; g < 4; ++g) {
            const int n = g * 256 + ji * 16 + ln;  // B column = Wx row
            const _Float16* wrow = Wx + (size_t)n * IN_DIM + k0 + half * 8;
            v8h b0 = *reinterpret_cast<const v8h*>(wrow);
            v8h b1 = *reinterpret_cast<const v8h*>(wrow + 16);
            v16h b;
            #pragma unroll
            for (int e = 0; e < 8; ++e) { b[e] = b0[e]; b[8 + e] = b1[e]; }
            acc[g] = __builtin_amdgcn_wmma_f32_16x16x32_f16(
                false, a, false, b, (short)0, acc[g], false, false);
        }
    }

    // bias + store pre-swizzled
    const int j = ji * 16 + ln;
    const float bb[4] = { bf[j], bi[j], bg[j], bo[j] };
    const int t  = m_tile >> 4;
    const int mi = m_tile & 15;
    #pragma unroll
    for (int g = 0; g < 4; ++g) {
        #pragma unroll
        for (int r = 0; r < 8; ++r) acc[g][r] += bb[g];
        size_t patch = ((size_t)(t * 4 + g) * 16 + ji) * 16 + mi;
        *reinterpret_cast<v8f*>(xz + patch * 256 + lane * 8) = acc[g];
    }
}

// ---------------------------------------------------------------------------
// Phase B: persistent recurrence. 32 blocks x 256 threads (8 waves).
// wave (mi, ji) owns batch rows mi*16.. and hidden cols ji*16.. ; per step:
//   4 gate tiles via 8 K-steps of v_wmma_f32_16x16x32_f16 (B from LDS),
//   z=-sin(xz+hz), gates, c/h update (c in registers), h -> out + f16 hbuf.
// ---------------------------------------------------------------------------
__global__ __launch_bounds__(256)
void qlstm_recur(const _Float16* __restrict__ Wh,
                 const float* __restrict__ xz,
                 _Float16* __restrict__ hbuf,      // [2][256*256] f16
                 float* __restrict__ out,          // outputs | hx | cx
                 unsigned* __restrict__ ctr,
                 int nblocks)
{
    __shared__ __align__(16) _Float16 lds[4 * 16 * 264];  // Wh slice, padded pitch

    const int tid  = threadIdx.x;
    const int lane = tid & 31;
    const int wi   = tid >> 5;
    const int ji   = blockIdx.x >> 1;                 // hidden tile (shared per block)
    const int mi   = ((blockIdx.x & 1) << 3) + wi;    // batch tile per wave
    const int j0   = ji * 16;
    const int m0   = mi * 16;
    const int half = lane >> 4;
    const int ln   = lane & 15;

    // preload this block's Wh slice: [4 gates][16 rows][256 K] -> LDS (pitch 264)
    for (int idx = tid; idx < 4 * 16 * HID; idx += 256) {
        int g = idx >> 12;
        int r = (idx >> 8) & 15;
        int k = idx & 255;
        lds[(g * 16 + r) * 264 + k] = Wh[(size_t)(g * 256 + j0 + r) * HID + k];
    }
    __syncthreads();

    v8f c = {};     // cell state patch lives in registers for all 512 steps

    // prefetch step 0's x-projection (coalesced 32B/lane)
    v8f xv[4];
    #pragma unroll
    for (int g = 0; g < 4; ++g) {
        size_t patch = ((size_t)g * 16 + ji) * 16 + mi;
        xv[g] = *reinterpret_cast<const v8f*>(xz + patch * 256 + lane * 8);
    }

    for (int t = 0; t < SEQ; ++t) {
        v8f acc[4] = { {}, {}, {}, {} };

        if (t > 0) {  // h-projection GEMM (h_prev == 0 at t==0)
            const _Float16* arow =
                hbuf + (size_t)((t - 1) & 1) * (BATCH * HID) + (size_t)(m0 + ln) * HID;
            #pragma unroll 2
            for (int k0 = 0; k0 < HID; k0 += 32) {
                v8h a0 = *reinterpret_cast<const v8h*>(arow + k0 + half * 8);
                v8h a1 = *reinterpret_cast<const v8h*>(arow + k0 + 16 + half * 8);
                v16h a;
                #pragma unroll
                for (int e = 0; e < 8; ++e) { a[e] = a0[e]; a[8 + e] = a1[e]; }
                #pragma unroll
                for (int g = 0; g < 4; ++g) {
                    const _Float16* lrow = lds + (g * 16 + ln) * 264 + k0 + half * 8;
                    v8h b0 = *reinterpret_cast<const v8h*>(lrow);
                    v8h b1 = *reinterpret_cast<const v8h*>(lrow + 16);
                    v16h b;
                    #pragma unroll
                    for (int e = 0; e < 8; ++e) { b[e] = b0[e]; b[8 + e] = b1[e]; }
                    acc[g] = __builtin_amdgcn_wmma_f32_16x16x32_f16(
                        false, a, false, b, (short)0, acc[g], false, false);
                }
            }
        }

        // add prefetched x-projection
        #pragma unroll
        for (int g = 0; g < 4; ++g) acc[g] += xv[g];

        // quantum encoder closed form + LSTM cell (elementwise, WMMA C/D layout)
        v8f h;
        #pragma unroll
        for (int r = 0; r < 8; ++r) {
            float zf = -__sinf(acc[0][r]);
            float zi = -__sinf(acc[1][r]);
            float zg = -__sinf(acc[2][r]);
            float zo = -__sinf(acc[3][r]);
            float fg = sigm(zf);
            float ig = sigm(zi);
            float gg = ftanh(zg);
            float og = sigm(zo);
            c[r] = fg * c[r] + ig * gg;
            h[r] = og * ftanh(c[r]);
        }

        // prefetch NEXT step's xz now: its HBM latency hides under the
        // stores + device-wide barrier below (registers survive the fences)
        v8f xn[4] = { {}, {}, {}, {} };
        if (t + 1 < SEQ) {
            #pragma unroll
            for (int g = 0; g < 4; ++g) {
                size_t patch = ((size_t)((t + 1) * 4 + g) * 16 + ji) * 16 + mi;
                xn[g] = *reinterpret_cast<const v8f*>(xz + patch * 256 + lane * 8);
            }
        }

        // write h: fp32 outputs + f16 ping-pong buffer for next step
        _Float16* hcur = hbuf + (size_t)(t & 1) * (BATCH * HID);
        #pragma unroll
        for (int r = 0; r < 8; ++r) {
            int m = m0 + r + half * 8;
            int j = j0 + ln;
            out[((size_t)t * BATCH + m) * HID + j] = h[r];
            hcur[(size_t)m * HID + j] = (_Float16)h[r];
        }
        if (t == SEQ - 1) {
            size_t hx_off = (size_t)SEQ * BATCH * HID;
            #pragma unroll
            for (int r = 0; r < 8; ++r) {
                int m = m0 + r + half * 8;
                int j = j0 + ln;
                out[hx_off + (size_t)m * HID + j]               = h[r];
                out[hx_off + BATCH * HID + (size_t)m * HID + j] = c[r];
            }
        } else {
            // device-wide barrier (monotone counter: no reset races)
            __threadfence();     // ALL threads: release my h stores to device scope
            __syncthreads();     // whole block's stores are now flushed
            if (tid == 0) {
                atomicAdd(ctr, 1u);
                unsigned target = (unsigned)nblocks * (unsigned)(t + 1);
                while (atomicAdd(ctr, 0u) < target) {
                    __builtin_amdgcn_s_sleep(2);
                }
            }
            __syncthreads();
            __threadfence();     // acquire: drop stale near-cache lines of hbuf
        }

        #pragma unroll
        for (int g = 0; g < 4; ++g) xv[g] = xn[g];
    }
}

// ---------------------------------------------------------------------------
extern "C" void kernel_launch(void* const* d_in, const int* in_sizes, int n_in,
                              void* d_out, int out_size, void* d_ws, size_t ws_size,
                              hipStream_t stream) {
    const float* x  = (const float*)d_in[0];
    const float* Wf = (const float*)d_in[1];
    const float* bf = (const float*)d_in[2];
    const float* Wi = (const float*)d_in[3];
    const float* bi = (const float*)d_in[4];
    const float* Wg = (const float*)d_in[5];
    const float* bg = (const float*)d_in[6];
    const float* Wo = (const float*)d_in[7];
    const float* bo = (const float*)d_in[8];

    char* ws = (char*)d_ws;
    unsigned* ctr  = (unsigned*)(ws + WS_CTR);
    _Float16* Wx   = (_Float16*)(ws + WS_WX);
    _Float16* Wh   = (_Float16*)(ws + WS_WH);
    _Float16* hbuf = (_Float16*)(ws + WS_HBUF);
    float*    xz   = (float*)(ws + WS_XZ);
    float*    out  = (float*)d_out;

    // zero the barrier counter every launch (graph-capturable memset node)
    hipMemsetAsync(ctr, 0, 256, stream);

    // Phase 0: weight split + f32->f16
    qlstm_convw<<<(1024 * 768 + 255) / 256, 256, 0, stream>>>(Wf, Wi, Wg, Wo, Wx, Wh);

    // Phase A: big parallel WMMA GEMM (131072 waves)
    qlstm_xproj<<<16384, 256, 0, stream>>>(x, Wx, bf, bi, bg, bo, xz);

    // Phase B: persistent sequential recurrence (32 blocks, all resident)
    qlstm_recur<<<32, 256, 0, stream>>>(Wh, xz, hbuf, out, ctr, 32);
}
